// Attention_6614249636620
// MI455X (gfx1250) — matrix operand order repaired and checked by reading, not compiled, not run on hardware
//
#include <hip/hip_runtime.h>
#include <math.h>

// ---------------------------------------------------------------------------
// Problem constants (from reference): B=4, S=4096, F=33, D=64, TEMPERATURE=1
// out = softmax( (q k^T) / rowmax(q k^T) ), q = xWq+bq, k = xWk+bk
// ---------------------------------------------------------------------------
#define BB 4
#define SS 4096
#define FF 33
#define DD 64

typedef __attribute__((ext_vector_type(16))) __bf16 v16bf;
typedef __attribute__((ext_vector_type(8)))  float  v8f;

// ---------------------------------------------------------------------------
// Kernel 1: fp32 projections -> split-precision bf16 (hi + lo) q and k.
// 4 rows per 256-thread block (64 threads per row, one d each).
// ---------------------------------------------------------------------------
__global__ void proj_kernel(const float* __restrict__ x,
                            const float* __restrict__ Wq, const float* __restrict__ bq,
                            const float* __restrict__ Wk, const float* __restrict__ bk,
                            __bf16* __restrict__ qh, __bf16* __restrict__ ql,
                            __bf16* __restrict__ kh, __bf16* __restrict__ kl) {
    __shared__ float xs[4][FF];
    const int tid = threadIdx.x;
    const int row = blockIdx.x * 4 + (tid >> 6);   // global row in [0, B*S)
    const int d   = tid & 63;

    if (tid < 4 * FF) {
        int r = tid / FF, c = tid % FF;
        xs[r][c] = x[(size_t)(blockIdx.x * 4 + r) * FF + c];
    }
    __syncthreads();

    const float* xr = xs[tid >> 6];
    float accq = bq[d], acck = bk[d];
#pragma unroll
    for (int f = 0; f < FF; ++f) {
        float xv = xr[f];
        accq = fmaf(xv, Wq[f * DD + d], accq);
        acck = fmaf(xv, Wk[f * DD + d], acck);
    }
    size_t o = (size_t)row * DD + d;
    __bf16 h;
    h = (__bf16)accq;  qh[o] = h;  ql[o] = (__bf16)(accq - (float)h);
    h = (__bf16)acck;  kh[o] = h;  kl[o] = (__bf16)(acck - (float)h);
}

// ---------------------------------------------------------------------------
// Kernel 2: per (batch, 16-row stripe): scores into 256KB LDS via bf16x3 WMMA,
// then rowmax-normalized softmax, then coalesced b128 writes.
// ---------------------------------------------------------------------------
union Frag { v16bf v; uint4 q2[2]; };

__global__ void attn_kernel(const __bf16* __restrict__ qh, const __bf16* __restrict__ ql,
                            const __bf16* __restrict__ kh, const __bf16* __restrict__ kl,
                            float* __restrict__ out) {
    extern __shared__ float sc[];                    // [16][SS] f32 = 256 KB (CDNA5 320KB LDS)

    const int b    = blockIdx.x / (SS / 16);
    const int m0   = (blockIdx.x % (SS / 16)) * 16;  // stripe of 16 rows
    const int wave = threadIdx.x >> 5;               // 8 waves of 32 (wave32)
    const int lane = threadIdx.x & 31;
    const int hl   = lane >> 4;                      // lane half (ISA A/B layout)
    const int l16  = lane & 15;

    const size_t qbase = ((size_t)b * SS + m0) * DD;
    const size_t kbase = (size_t)b * SS * DD;

    // A fragments (q stripe), held in registers for the whole loop.
    // 16-bit A 16x32 layout: lane row = lane&15; elements 0..7 = K hl*8..hl*8+7,
    // elements 8..15 = K 16+hl*8..16+hl*8+7  (two 16B contiguous chunks).
    Frag Ah[2], Al[2];
#pragma unroll
    for (int c = 0; c < 2; ++c) {                    // K-chunks: d 0..31 and 32..63
        size_t base = qbase + (size_t)l16 * DD + c * 32 + hl * 8;
        Ah[c].q2[0] = *(const uint4*)(qh + base);
        Ah[c].q2[1] = *(const uint4*)(qh + base + 16);
        Al[c].q2[0] = *(const uint4*)(ql + base);
        Al[c].q2[1] = *(const uint4*)(ql + base + 16);
    }

    // Each wave owns column tiles j = wave, wave+8, ... (32 tiles each).
    for (int j = wave; j < SS / 16; j += 8) {
        const int n0 = j * 16;
        // B 32x16 layout: lane column = lane&15 (k-row n0+l16); elements 0..7 =
        // K c*32+hl*8..+7, elements 8..15 = K c*32+16+hl*8..+7 (two b128 loads).
        Frag Bh[2], Bl[2];
        size_t krow = kbase + (size_t)(n0 + l16) * DD;
#pragma unroll
        for (int c = 0; c < 2; ++c) {
            size_t base = krow + c * 32 + hl * 8;
            Bh[c].q2[0] = *(const uint4*)(kh + base);
            Bh[c].q2[1] = *(const uint4*)(kh + base + 16);
            Bl[c].q2[0] = *(const uint4*)(kl + base);
            Bl[c].q2[1] = *(const uint4*)(kl + base + 16);
        }

        // bf16x3 split accumulation in f32: qh*kh + ql*kh + qh*kl  (ql*kl dropped).
        // Two independent accumulators (one per K-chunk) break the WMMA->WMMA
        // D->C RAW chain so the scheduler can alternate them without v_nops
        // (WMMA_*BF16 producer->consumer needs ~5 slots per ISA 7.12.1).
        v8f acc0 = {}, acc1 = {};
        acc0 = __builtin_amdgcn_wmma_f32_16x16x32_bf16(false, Ah[0].v, false, Bh[0].v, (short)0, acc0, false, false);
        acc1 = __builtin_amdgcn_wmma_f32_16x16x32_bf16(false, Ah[1].v, false, Bh[1].v, (short)0, acc1, false, false);
        acc0 = __builtin_amdgcn_wmma_f32_16x16x32_bf16(false, Al[0].v, false, Bh[0].v, (short)0, acc0, false, false);
        acc1 = __builtin_amdgcn_wmma_f32_16x16x32_bf16(false, Al[1].v, false, Bh[1].v, (short)0, acc1, false, false);
        acc0 = __builtin_amdgcn_wmma_f32_16x16x32_bf16(false, Ah[0].v, false, Bl[0].v, (short)0, acc0, false, false);
        acc1 = __builtin_amdgcn_wmma_f32_16x16x32_bf16(false, Ah[1].v, false, Bl[1].v, (short)0, acc1, false, false);

        // C/D layout: VGPR r -> row r + 8*hl, col = lane&15.
#pragma unroll
        for (int r = 0; r < 8; ++r)
            sc[(size_t)(r + 8 * hl) * SS + n0 + l16] = acc0[r] + acc1[r];
    }
    __syncthreads();

    // Softmax over t = s / rowmax(s). Note: max(t) = 1 if rowmax>0 else rowmin/rowmax,
    // so track both max and min. Each wave owns 2 rows; lanes stride in float4.
    for (int rr = 0; rr < 2; ++rr) {
        const int m = wave * 2 + rr;
        float4* rowp = (float4*)(sc + (size_t)m * SS);

        float mx = -__builtin_inff(), mn = __builtin_inff();
        for (int t = lane; t < SS / 4; t += 32) {
            float4 v = rowp[t];
            mx = fmaxf(fmaxf(mx, v.x), fmaxf(fmaxf(v.y, v.z), v.w));
            mn = fminf(fminf(mn, v.x), fminf(fminf(v.y, v.z), v.w));
        }
#pragma unroll
        for (int off = 16; off > 0; off >>= 1) {
            mx = fmaxf(mx, __shfl_xor(mx, off, 32));
            mn = fminf(mn, __shfl_xor(mn, off, 32));
        }
        const float inv_m = 1.0f / mx;
        const float m2 = (mx > 0.0f) ? 1.0f : mn * inv_m;   // max of s/mx

        float sum = 0.0f;
        for (int t = lane; t < SS / 4; t += 32) {
            float4 v = rowp[t];
            v.x = __expf(fmaf(v.x, inv_m, -m2));
            v.y = __expf(fmaf(v.y, inv_m, -m2));
            v.z = __expf(fmaf(v.z, inv_m, -m2));
            v.w = __expf(fmaf(v.w, inv_m, -m2));
            rowp[t] = v;
            sum += (v.x + v.y) + (v.z + v.w);
        }
#pragma unroll
        for (int off = 16; off > 0; off >>= 1)
            sum += __shfl_xor(sum, off, 32);
        const float inv_s = 1.0f / sum;

        float4* outp = (float4*)(out + ((size_t)(b * SS + m0 + m)) * SS);
        for (int t = lane; t < SS / 4; t += 32) {
            float4 v = rowp[t];
            v.x *= inv_s; v.y *= inv_s; v.z *= inv_s; v.w *= inv_s;
            outp[t] = v;                                   // coalesced b128 stores
        }
    }
}

// ---------------------------------------------------------------------------
extern "C" void kernel_launch(void* const* d_in, const int* in_sizes, int n_in,
                              void* d_out, int out_size, void* d_ws, size_t ws_size,
                              hipStream_t stream) {
    const float* x  = (const float*)d_in[0];
    const float* Wq = (const float*)d_in[1];
    const float* bq = (const float*)d_in[2];
    const float* Wk = (const float*)d_in[3];
    const float* bk = (const float*)d_in[4];
    float* out = (float*)d_out;

    const size_t n = (size_t)BB * SS * DD;             // 1M elements per array
    __bf16* qh = (__bf16*)d_ws;                        // 4 bf16 arrays = 8 MB total
    __bf16* ql = qh + n;
    __bf16* kh = ql + n;
    __bf16* kl = kh + n;

    proj_kernel<<<BB * SS / 4, 256, 0, stream>>>(x, Wq, bq, Wk, bk, qh, ql, kh, kl);
    attn_kernel<<<BB * (SS / 16), 256, 16 * SS * sizeof(float), stream>>>(qh, ql, kh, kl, out);
}
